// CorrelateAttention_90640989815219
// MI455X (gfx1250) — matrix operand order repaired and checked by reading, not compiled, not run on hardware
//
#include <hip/hip_runtime.h>
#include <math.h>

// ---------------- problem constants ----------------
#define HID     2048
#define OUTD    2560
#define QSZ     2048
#define NHEADS  16
#define NKHEADS 4
#define HDIM    128
#define BATCH   4
#define SEQ     2048
#define MROWS   (BATCH * SEQ)   // 8192

typedef __attribute__((ext_vector_type(16))) __bf16 v16bf;
typedef __attribute__((ext_vector_type(8)))  __bf16 v8bf;
typedef __attribute__((ext_vector_type(8)))  float  v8f;

__device__ __forceinline__ unsigned short f32_to_bf16(float f) {
  unsigned int u = __float_as_uint(f);
  u += 0x7FFFu + ((u >> 16) & 1u);      // round-to-nearest-even
  return (unsigned short)(u >> 16);
}

// A fragment (16x32 bf16). Lane = m (0..15) | half<<4.
// elements e -> K = half*16 + e (contiguous).
// p must already include (+ half*16) in elements.
__device__ __forceinline__ v16bf load_fragA(const unsigned short* p) {
  v8bf lo = *reinterpret_cast<const v8bf*>(p);
  v8bf hi = *reinterpret_cast<const v8bf*>(p + 8);
  return __builtin_shufflevector(lo, hi, 0,1,2,3,4,5,6,7,8,9,10,11,12,13,14,15);
}

// B fragment (32x16 bf16). Lane = n (0..15) | half<<4.
// elements e -> K = half*16 + e (contiguous 32 bytes per lane, 32B aligned).
__device__ __forceinline__ v16bf load_fragB(const unsigned short* p) {
  return *reinterpret_cast<const v16bf*>(p);
}

#define WMMA_BF16(A, B, C)                                             \
  __builtin_amdgcn_wmma_f32_16x16x32_bf16(false, (A), false, (B),      \
                                          (short)0, (C), false, false)

// ---------------- kernel 1: f32 -> bf16 conversion ----------------
__global__ void cvt_bf16_kernel(const float4* __restrict__ in,
                                ushort4* __restrict__ out, int n4) {
  int i = blockIdx.x * blockDim.x + threadIdx.x;
  int stride = gridDim.x * blockDim.x;
  for (; i < n4; i += stride) {
    float4 v = in[i];
    ushort4 o;
    o.x = f32_to_bf16(v.x); o.y = f32_to_bf16(v.y);
    o.z = f32_to_bf16(v.z); o.w = f32_to_bf16(v.w);
    out[i] = o;
  }
}

// ---------------- kernel 2: QK projection GEMM (bf16 WMMA) ----------------
// C[8192,2560] = H[8192,2048] * W[2560,2048]^T + bias, fused epilogue:
//   n <  2048 : q, scaled by R_SOFTPLUS_0/128 * softplus(scaling[d]) -> qb[B,H,L,D]
//   n >= 2048 : k -> kb[B,Kh,L,D]
// 256 threads = 8 waves arranged 2(M) x 4(N); wave tile 32x32 (4 accumulators,
// 2 A-frags + 2 B-frags per K-step -> 2 b128 loads per WMMA). Block tile 64x128.
__global__ __launch_bounds__(256) void qk_proj_gemm(
    const unsigned short* __restrict__ Hb, const unsigned short* __restrict__ Wb,
    const float* __restrict__ bias, const float* __restrict__ scaling,
    unsigned short* __restrict__ qb, unsigned short* __restrict__ kb) {
  const int t    = threadIdx.x & 31;
  const int wave = threadIdx.x >> 5;
  const int Mbase = blockIdx.x * 64  + (wave >> 2) * 32;
  const int Nbase = blockIdx.y * 128 + (wave & 3) * 32;
  const int m = t & 15, half = t >> 4;

  const unsigned short* a0row = Hb + (size_t)(Mbase + m) * HID + half * 16;
  const unsigned short* a1row = a0row + (size_t)16 * HID;
  const unsigned short* b0row = Wb + (size_t)(Nbase + m) * HID + half * 16;
  const unsigned short* b1row = b0row + (size_t)16 * HID;

  v8f acc[2][2];
  #pragma unroll
  for (int i = 0; i < 2; ++i)
    #pragma unroll
    for (int j = 0; j < 2; ++j) acc[i][j] = (v8f){};

  #pragma unroll 2
  for (int k0 = 0; k0 < HID; k0 += 32) {
    __builtin_prefetch(a0row + k0 + 256, 0, 1);   // global_prefetch_b8
    __builtin_prefetch(b0row + k0 + 256, 0, 1);
    v16bf a0 = load_fragA(a0row + k0);
    v16bf a1 = load_fragA(a1row + k0);
    v16bf b0 = load_fragB(b0row + k0);
    v16bf b1 = load_fragB(b1row + k0);
    acc[0][0] = WMMA_BF16(a0, b0, acc[0][0]);
    acc[0][1] = WMMA_BF16(a0, b1, acc[0][1]);
    acc[1][0] = WMMA_BF16(a1, b0, acc[1][0]);
    acc[1][1] = WMMA_BF16(a1, b1, acc[1][1]);
  }

  // Epilogue. C/D layout: element e -> M = tileM + half*8 + e, N = lane&15.
  #pragma unroll
  for (int i = 0; i < 2; ++i) {
    #pragma unroll
    for (int j = 0; j < 2; ++j) {
      v8f c = acc[i][j];
      const int gn = Nbase + j * 16 + m;
      const float bv = bias[gn];
      const int d = gn & (HDIM - 1);
      const bool isq = (gn < QSZ);         // uniform per accumulator per wave
      float sc = 1.0f;
      if (isq) {
        float x = scaling[d];
        float sp = (x > 15.0f) ? x : __logf(1.0f + __expf(x));
        // folds R_SOFTPLUS_0/sqrt(D) * softplus AND the later 1/sqrt(D)
        sc = 1.442695041f * sp * (1.0f / 128.0f);
      }
      #pragma unroll
      for (int e = 0; e < 8; ++e) {
        const int gm = Mbase + i * 16 + half * 8 + e;
        const int bI = gm >> 11;           // / SEQ
        const int l  = gm & (SEQ - 1);
        float v = c[e] + bv;
        if (isq) {
          const int h = gn >> 7;
          qb[(((size_t)(bI * NHEADS + h) * SEQ) + l) * HDIM + d] =
              f32_to_bf16(v * sc);
        } else {
          const int kh = (gn - QSZ) >> 7;
          kb[(((size_t)(bI * NKHEADS + kh) * SEQ) + l) * HDIM + d] =
              f32_to_bf16(v);
        }
      }
    }
  }
}

// ---------------- kernel 3: attention scores + softmax + head-mean ----------
// One workgroup per (batch, 16-row query tile). LDS: logits[16][2048] f32 +
// acc[16][2048] f32 (256 KB of the 320 KB WGP LDS). Softmax row reductions use
// __shfl_xor within 16-lane groups (each row group lives inside one wave), so
// only 2 barriers per head are needed.
__global__ __launch_bounds__(256) void attn_softmax_mean(
    const unsigned short* __restrict__ qb, const unsigned short* __restrict__ kb,
    float* __restrict__ out) {
  extern __shared__ float smem[];
  float* logits = smem;             // 16*SEQ
  float* acc    = smem + 16 * SEQ;  // 16*SEQ

  const int b   = blockIdx.x >> 7;
  const int i0  = (blockIdx.x & 127) << 4;
  const int t    = threadIdx.x & 31;
  const int w    = threadIdx.x >> 5;
  const int m    = t & 15, half = t >> 4;
  const int row  = threadIdx.x >> 4;   // 0..15 (16-thread group within a wave)
  const int sub  = threadIdx.x & 15;   // 0..15

  for (int i = threadIdx.x; i < 16 * SEQ; i += 256) acc[i] = 0.0f;
  __syncthreads();

  for (int h = 0; h < NHEADS; ++h) {
    const int kh = h >> 2;  // GQA: 4 q-heads share one k-head
    const unsigned short* qrow =
        qb + (((size_t)(b * NHEADS + h) * SEQ) + (i0 + m)) * HDIM + half * 16;
    const unsigned short* kbh =
        kb + ((size_t)(b * NKHEADS + kh) * SEQ) * HDIM + half * 16;

    // Each of the 8 waves computes 16 j-tiles of 16 columns (128 tiles total).
    // Q fragments are loop-invariant and get hoisted; per tile: 8 b128 loads,
    // 4 WMMAs.
    #pragma unroll 2
    for (int s = 0; s < 16; ++s) {
      const int j0 = (w + 8 * s) << 4;
      const unsigned short* krow = kbh + (size_t)(j0 + m) * HDIM;
      v8f c = {};
      #pragma unroll
      for (int dk = 0; dk < HDIM; dk += 32) {
        v16bf a  = load_fragA(qrow + dk);
        v16bf bb = load_fragB(krow + dk);
        c = WMMA_BF16(a, bb, c);
      }
      #pragma unroll
      for (int e = 0; e < 8; ++e)
        logits[(half * 8 + e) * SEQ + j0 + m] = c[e];  // 1/sqrt(D) already in q
    }
    __syncthreads();  // all waves' logits visible

    // Row-wise softmax: 16 threads per row, wave-local shuffle reductions.
    float* lrow = logits + row * SEQ;
    float lmax = -3.402823466e38f;
    for (int j = sub; j < SEQ; j += 16) lmax = fmaxf(lmax, lrow[j]);
    #pragma unroll
    for (int msk = 8; msk >= 1; msk >>= 1)
      lmax = fmaxf(lmax, __shfl_xor(lmax, msk, 16));

    float lsum = 0.0f;
    for (int j = sub; j < SEQ; j += 16) {
      float e2 = __expf(lrow[j] - lmax);
      lrow[j] = e2;
      lsum += e2;
    }
    #pragma unroll
    for (int msk = 8; msk >= 1; msk >>= 1)
      lsum += __shfl_xor(lsum, msk, 16);

    const float rinv = 1.0f / (lsum * (float)NHEADS);  // mean over 16 heads
    for (int j = sub; j < SEQ; j += 16) acc[row * SEQ + j] += lrow[j] * rinv;
    __syncthreads();  // logits reused by next head
  }

  float* obase = out + ((size_t)b * SEQ + i0) * SEQ;
  for (int i = threadIdx.x; i < 16 * SEQ; i += 256) obase[i] = acc[i];
}

// ---------------- launcher ----------------
extern "C" void kernel_launch(void* const* d_in, const int* in_sizes, int n_in,
                              void* d_out, int out_size, void* d_ws, size_t ws_size,
                              hipStream_t stream) {
  const float* hs   = (const float*)d_in[0];  // [4,2048,2048]
  const float* wqk  = (const float*)d_in[1];  // [2560,2048]
  const float* bias = (const float*)d_in[2];  // [2560]
  const float* scal = (const float*)d_in[3];  // [128]
  float* out = (float*)d_out;                 // [4,2048,2048]

  char* ws = (char*)d_ws;
  size_t offH = 0;
  size_t offW = offH + (size_t)MROWS * HID * 2;                    // 32 MB
  size_t offQ = offW + (size_t)OUTD * HID * 2;                     // +10 MB
  size_t offK = offQ + (size_t)BATCH * NHEADS * SEQ * HDIM * 2;    // +32 MB
  unsigned short* Hb = (unsigned short*)(ws + offH);
  unsigned short* Wb = (unsigned short*)(ws + offW);
  unsigned short* Qb = (unsigned short*)(ws + offQ);
  unsigned short* Kb = (unsigned short*)(ws + offK);               // +8 MB

  cvt_bf16_kernel<<<dim3(2048), dim3(256), 0, stream>>>(
      (const float4*)hs, (ushort4*)Hb, MROWS * HID / 4);
  cvt_bf16_kernel<<<dim3(2048), dim3(256), 0, stream>>>(
      (const float4*)wqk, (ushort4*)Wb, OUTD * HID / 4);

  qk_proj_gemm<<<dim3(MROWS / 64, OUTD / 128), dim3(256), 0, stream>>>(
      Hb, Wb, bias, scal, Qb, Kb);

  const size_t smem = (size_t)(32 * SEQ) * sizeof(float);  // 256 KB
  hipFuncSetAttribute((const void*)attn_softmax_mean,
                      hipFuncAttributeMaxDynamicSharedMemorySize, (int)smem);
  attn_softmax_mean<<<dim3(BATCH * (SEQ / 16)), dim3(256), smem, stream>>>(
      Qb, Kb, out);

  (void)in_sizes; (void)n_in; (void)out_size; (void)ws_size;
}
